// WanSelfAttention_19722489823620
// MI455X (gfx1250) — matrix operand order repaired and checked by reading, not compiled, not run on hardware
//
#include <hip/hip_runtime.h>

// ---------------------------------------------------------------------------
// Types / WMMA helpers (CDNA5 gfx1250, wave32)
// ---------------------------------------------------------------------------
typedef _Float16 h16;
typedef __attribute__((ext_vector_type(16))) _Float16 v16h;
typedef __attribute__((ext_vector_type(8)))  _Float16 v8h;
typedef __attribute__((ext_vector_type(8)))  float    v8f;
typedef __attribute__((ext_vector_type(4)))  unsigned u32x4;
typedef __attribute__((ext_vector_type(8)))  int      i32x8;
typedef __attribute__((ext_vector_type(4)))  int      i32x4;

#define SEQ 6912
#define DIM 1536
#define NH  12
#define HD  128
#define FB  12     // frames (a / f blocks)
#define BB  24     // block edge (i/j/k/l)
#define FK  288    // FB*BB (flattened f*k softmax width)

#define WMMA_F16(a, b, c) \
  __builtin_amdgcn_wmma_f32_16x16x32_f16(false, (a), false, (b), (short)0, (c), false, false)

// A fragment: 16x32 f16 tile, row-major with leading dim `ld` (halfs).
// Lane L (<16): m=L, K halfs {0..7, 16..23}; lane L (>=16): m=L-16, K {8..15, 24..31}.
__device__ __forceinline__ v16h frag_a(const h16* t, int ld, int lane) {
  int m  = lane & 15;
  int kb = (lane >> 4) << 3;
  const h16* p = t + (size_t)m * ld + kb;
  union { v16h v; v8h h[2]; } f;
  f.h[0] = *(const v8h*)(p);
  f.h[1] = *(const v8h*)(p + 16);
  return f.v;
}

// B fragment: 32x16 f16 tile stored transposed (element B[k][n] at t[n*ld + k]).
// Lane L (<16): n=L, K {0..7, 16..23}; lane L (>=16): n=L-16, K {8..15, 24..31}.
__device__ __forceinline__ v16h frag_b(const h16* t, int ld, int lane) {
  int n  = lane & 15;
  int kb = (lane >> 4) << 3;
  const h16* p = t + (size_t)n * ld + kb;
  union { v16h v; v8h h[2]; } f;
  f.h[0] = *(const v8h*)(p);
  f.h[1] = *(const v8h*)(p + 16);
  return f.v;
}

// ---------------------------------------------------------------------------
// Tensor Data Mover: DMA a 2D f16 tile (tile_rows x tile_cols, row stride
// row_stride_elems) from global memory into LDS at byte offset lds_byte_off.
// D# packed per CDNA5 ISA 8.3-8.5: group0 {count=1, lds_addr, global_addr,
// type=2}; group1 {data_size=2B, tensor_dim0/1, tile_dim0/1, dim0 stride};
// remaining groups zero (2D tensor). Tracked by TENSORcnt.
// This toolchain exposes the 6-arg builtin (extra int32x8 group before cpol).
// ---------------------------------------------------------------------------
__device__ __forceinline__ void tdm_load_2d_f16(
    unsigned lds_byte_off, const h16* gptr, int tile_rows, int tile_cols,
    int row_stride_elems, int tensor_rows) {
  unsigned long long ga = (unsigned long long)(const void*)gptr;
  u32x4 g0;
  g0[0] = 1u;                                                  // count=1
  g0[1] = lds_byte_off;                                        // lds_addr
  g0[2] = (unsigned)ga;                                        // global_addr[31:0]
  g0[3] = (unsigned)((ga >> 32) & 0x01ffffffu) | (2u << 30);   // addr[56:32] | type=2
  i32x8 g1;
  g1[0] = 1 << 16;                                             // data_size = 1 -> 2 bytes
  g1[1] = (row_stride_elems & 0xffff) << 16;                   // tensor_dim0[15:0]
  g1[2] = ((row_stride_elems >> 16) & 0xffff) | ((tensor_rows & 0xffff) << 16);
  g1[3] = ((tensor_rows >> 16) & 0xffff) | (tile_cols << 16);  // tensor_dim1 hi | tile_dim0
  g1[4] = tile_rows & 0xffff;                                  // tile_dim1 (tile_dim2=0)
  g1[5] = row_stride_elems;                                    // tensor_dim0_stride[31:0]
  g1[6] = 0;
  g1[7] = 0;
  i32x4 gz4 = {0, 0, 0, 0};
  i32x8 gz8 = {0, 0, 0, 0, 0, 0, 0, 0};
  __builtin_amdgcn_tensor_load_to_lds(g0, g1, gz4, gz4, gz8, 0);
}

// ---------------------------------------------------------------------------
// Kernel 0: f32 -> f16 conversion (one-shot; halves GEMM streaming traffic)
// ---------------------------------------------------------------------------
__global__ __launch_bounds__(256) void to_f16(
    const float* __restrict__ src, h16* __restrict__ dst, unsigned long long n) {
  unsigned long long i = (unsigned long long)blockIdx.x * 256 + threadIdx.x;
  unsigned long long stride = (unsigned long long)gridDim.x * 256;
  for (; i < n; i += stride) dst[i] = (h16)src[i];
}

// ---------------------------------------------------------------------------
// Kernel 1/7: C[M,N] = A[M,K] @ W[N,K]^T + bias (all-f16 operands, f32 accum).
// Block = 256 threads = 8 waves (4x2), tile 128x64x32, TDM double-buffered
// global->LDS tile staging (wave 0 issues, s_wait_tensorcnt retires in order).
// ---------------------------------------------------------------------------
__global__ __launch_bounds__(256) void gemm_f16(
    const h16* __restrict__ A, const h16* __restrict__ W,
    const float* __restrict__ bias, float* __restrict__ C,
    int M, int N, int K) {
  __shared__ h16 As[2][128][32];
  __shared__ h16 Bs[2][64][32];   // stored [n][k]: directly B-fragment friendly
  int tid = threadIdx.x, lane = tid & 31, wave = tid >> 5;
  int m0 = blockIdx.y * 128, n0 = blockIdx.x * 64;
  int wm = (wave >> 1) * 32, wn = (wave & 1) * 32;
  v8f zero = {};
  v8f acc[2][2];
  for (int mt = 0; mt < 2; ++mt)
    for (int nt = 0; nt < 2; ++nt) acc[mt][nt] = zero;

  // LDS flat aperture keeps the byte offset in addr[31:0].
  unsigned asb[2], bsb[2];
  asb[0] = (unsigned)(unsigned long long)(void*)&As[0][0][0];
  asb[1] = (unsigned)(unsigned long long)(void*)&As[1][0][0];
  bsb[0] = (unsigned)(unsigned long long)(void*)&Bs[0][0][0];
  bsb[1] = (unsigned)(unsigned long long)(void*)&Bs[1][0][0];

  const int nsteps = K >> 5;
  if (wave == 0) {
    tdm_load_2d_f16(asb[0], A + (size_t)m0 * K, 128, 32, K, M - m0);
    tdm_load_2d_f16(bsb[0], W + (size_t)n0 * K, 64, 32, K, N - n0);
  }
  for (int s = 0; s < nsteps; ++s) {
    int cur = s & 1;
    if (wave == 0) {
      if (s + 1 < nsteps) {
        tdm_load_2d_f16(asb[cur ^ 1], A + (size_t)m0 * K + (s + 1) * 32, 128, 32, K, M - m0);
        tdm_load_2d_f16(bsb[cur ^ 1], W + (size_t)n0 * K + (s + 1) * 32, 64, 32, K, N - n0);
        __builtin_amdgcn_s_wait_tensorcnt(2);  // in-order: current pair landed
      } else {
        __builtin_amdgcn_s_wait_tensorcnt(0);
      }
    }
    __syncthreads();  // publish LDS tiles to all waves
    for (int mt = 0; mt < 2; ++mt) {
      v16h af = frag_a(&As[cur][wm + mt * 16][0], 32, lane);
      for (int nt = 0; nt < 2; ++nt) {
        v16h bf = frag_b(&Bs[cur][wn + nt * 16][0], 32, lane);
        acc[mt][nt] = WMMA_F16(af, bf, acc[mt][nt]);
      }
    }
    __syncthreads();  // tiles consumed before next TDM overwrite
  }
  int hi = (lane >> 4) << 3, nn = lane & 15;
  for (int mt = 0; mt < 2; ++mt)
    for (int nt = 0; nt < 2; ++nt)
      for (int r = 0; r < 8; ++r) {
        int m = m0 + wm + mt * 16 + r + hi;
        int n = n0 + wn + nt * 16 + nn;
        C[(size_t)m * N + n] = acc[mt][nt][r] + bias[n];
      }
}

// ---------------------------------------------------------------------------
// Kernel 2: per-token RMS norm (+gain) + 3D RoPE -> f16. mode: 0=q, 1=k, 2=v.
// ---------------------------------------------------------------------------
__global__ __launch_bounds__(256) void norm_rope(
    const float* __restrict__ X, const float* __restrict__ g,
    const float* __restrict__ freqs, h16* __restrict__ out, int mode) {
  int s = blockIdx.x, tid = threadIdx.x;
  const float* row = X + (size_t)s * DIM;
  if (mode == 2) {
    for (int c = tid; c < DIM; c += 256) out[(size_t)s * DIM + c] = (h16)row[c];
    return;
  }
  __shared__ float red[256];
  __shared__ float rmss;
  float ss = 0.f;
  for (int c = tid; c < DIM; c += 256) { float v = row[c]; ss += v * v; }
  red[tid] = ss;
  __syncthreads();
  for (int off = 128; off > 0; off >>= 1) {
    if (tid < off) red[tid] += red[tid + off];
    __syncthreads();
  }
  if (tid == 0) rmss = rsqrtf(red[0] / (float)DIM + 1e-6f);
  __syncthreads();
  float rms = rmss;
  int a = s / (BB * BB), rr = s % (BB * BB), i = rr / BB, j = rr % BB;
  // 12 heads x 64 rotary pairs; c0=22 (frame), c1=21 (height), c1=21 (width)
  for (int p = tid; p < NH * 64; p += 256) {
    int h = p >> 6, t = p & 63;
    int pos = (t < 22) ? a : ((t < 43) ? i : j);
    float ang = freqs[pos * 64 + t];
    float cs = __cosf(ang), sn = __sinf(ang);
    int c = h * HD + 2 * t;
    float xr = row[c] * rms * g[c];
    float xi = row[c + 1] * rms * g[c + 1];
    out[(size_t)s * DIM + c]     = (h16)(xr * cs - xi * sn);
    out[(size_t)s * DIM + c + 1] = (h16)(xr * sn + xi * cs);
  }
}

// ---------------------------------------------------------------------------
// Kernel 3: right step, one block per (h,a,f,k).
//   bR = Q_blk @ K_blk^T (24x24, padded 32x32), masked softmax over l -> R, cL
//   aL = R @ K_blk (24x128) -> aLw[h][a][f][j][k][d]
//   Y  = R @ V_blk (24x128) -> Ytw[h][a][j][d][f*24+k]  (pre-transposed for K6)
// ---------------------------------------------------------------------------
__global__ __launch_bounds__(128) void right_step(
    const h16* __restrict__ qh, const h16* __restrict__ kh, const h16* __restrict__ vh,
    h16* __restrict__ aLw, float* __restrict__ cLw, h16* __restrict__ Ytw) {
  __shared__ h16 Qs[32][HD];     // rows j (pad 24..31 = 0)
  __shared__ h16 Ks[32][HD];     // rows l (pad = 0)
  __shared__ h16 KT[HD][32];     // [d][l], l pad = 0
  __shared__ h16 VT[HD][32];     // [d][l], l pad = 0
  __shared__ float bRs[24][32];
  __shared__ h16 Rs[32][32];     // zero everywhere except [j<24][l<24]
  int blk = blockIdx.x;
  int k = blk % BB; int t = blk / BB;
  int f = t % FB; t /= FB;
  int a = t % FB; int h = t / FB;
  int tid = threadIdx.x, lane = tid & 31, wave = tid >> 5;

  for (int idx = tid; idx < 8 * HD; idx += 128) {
    int r = 24 + (idx >> 7), c = idx & 127;
    Qs[r][c] = (h16)0.f; Ks[r][c] = (h16)0.f;
    int d = idx >> 3, l = 24 + (idx & 7);
    KT[d][l] = (h16)0.f; VT[d][l] = (h16)0.f;
  }
  for (int idx = tid; idx < 32 * 32; idx += 128) Rs[idx >> 5][idx & 31] = (h16)0.f;

  for (int idx = tid; idx < BB * HD; idx += 128) {
    int r = idx >> 7, d = idx & 127;
    Qs[r][d] = qh[((size_t)((a * BB + k) * BB + r)) * DIM + h * HD + d];
    size_t ksrc = ((size_t)((f * BB + k) * BB + r)) * DIM + h * HD + d;
    h16 kv = kh[ksrc];
    Ks[r][d] = kv; KT[d][r] = kv;
    VT[d][r] = vh[ksrc];
  }
  __syncthreads();

  {  // bR: wave -> one 16x16 tile of the 32x32 (padded) score matrix
    int mt = wave >> 1, nt = wave & 1;
    v8f c = {};
    for (int kk = 0; kk < 4; ++kk) {
      v16h af = frag_a(&Qs[mt * 16][kk * 32], HD, lane);
      v16h bf = frag_b(&Ks[nt * 16][kk * 32], HD, lane);
      c = WMMA_F16(af, bf, c);
    }
    int hi = (lane >> 4) << 3, nn = lane & 15;
    for (int r = 0; r < 8; ++r) {
      int m = mt * 16 + r + hi, n = nt * 16 + nn;
      if (m < 24) bRs[m][n] = c[r];
    }
  }
  __syncthreads();

  // softmax over l<24 with z = bR * sm_scale / (1+eps); also cL = E[z-logz]
  const float SCL = 0.08838834764831845f * (1.0f / (1.0f + 1e-6f));
  if (tid < 24) {
    int j = tid;
    float mx = -1e30f;
    for (int l = 0; l < 24; ++l) mx = fmaxf(mx, bRs[j][l] * SCL);
    float se = 0.f, sez = 0.f;
    for (int l = 0; l < 24; ++l) {
      float z = bRs[j][l] * SCL - mx;
      float e = __expf(z);
      bRs[j][l] = e;
      se += e; sez += e * z;
    }
    float inv = 1.f / se;
    for (int l = 0; l < 24; ++l) Rs[j][l] = (h16)(bRs[j][l] * inv);
    cLw[(((size_t)((h * FB + a) * FB + f) * BB + j)) * BB + k] = sez * inv - __logf(se);
  }
  __syncthreads();

  {  // aL = R @ K, Y = R @ V : 2(M) x 8(N) tiles, 4 per wave, single K step (l=32)
    int hi = (lane >> 4) << 3, nn = lane & 15;
    size_t aLbase = (((size_t)((h * FB + a) * FB + f)) * BB * BB) * HD;
    for (int q = 0; q < 4; ++q) {
      int ti = wave * 4 + q, mt = ti >> 3, nt = ti & 7;
      v16h af = frag_a(&Rs[mt * 16][0], 32, lane);
      v16h bk = frag_b(&KT[nt * 16][0], 32, lane);
      v16h bv = frag_b(&VT[nt * 16][0], 32, lane);
      v8f z = {};
      v8f aL = WMMA_F16(af, bk, z);
      v8f Y  = WMMA_F16(af, bv, z);
      for (int r = 0; r < 8; ++r) {
        int m = mt * 16 + r + hi, n = nt * 16 + nn;  // m=j, n=d
        if (m < 24) {
          aLw[aLbase + ((size_t)m * BB + k) * HD + n] = (h16)aL[r];
          Ytw[(((size_t)((h * FB + a) * BB + m) * HD + n)) * FK + f * BB + k] = (h16)Y[r];
        }
      }
    }
  }
}

// ---------------------------------------------------------------------------
// Kernel 4: bL = aL @ Q_j^T - cL, one block per (h,a,f,j).
// Fragments loaded straight from global memory (layouts already match).
// logits[h][a][j][i][f*24+k]
// ---------------------------------------------------------------------------
__global__ __launch_bounds__(128) void left_logits(
    const h16* __restrict__ aLw, const float* __restrict__ cLw,
    const h16* __restrict__ qh, float* __restrict__ logits) {
  int blk = blockIdx.x;
  int j = blk % BB; int t = blk / BB;
  int f = t % FB; t /= FB;
  int a = t % FB; int h = t / FB;
  int tid = threadIdx.x, lane = tid & 31, wave = tid >> 5;
  const h16* aLb = aLw + ((((size_t)((h * FB + a) * FB + f)) * BB + j)) * BB * HD;  // [k][d]
  const h16* qb  = qh + ((size_t)(a * BB * BB + j)) * DIM + h * HD;                 // row i stride 24*DIM
  const float* cLb = cLw + (((size_t)((h * FB + a) * FB + f) * BB + j)) * BB;
  int mt = wave >> 1, nt = wave & 1;
  v8f c = {};
  for (int kk = 0; kk < 4; ++kk) {
    v16h af = frag_a(aLb + (size_t)(mt * 16) * HD + kk * 32, HD, lane);
    v16h bf = frag_b(qb + (size_t)(nt * 16) * (BB * DIM) + kk * 32, BB * DIM, lane);
    c = WMMA_F16(af, bf, c);
  }
  size_t lb = (size_t)((h * FB + a) * BB + j);
  int hi = (lane >> 4) << 3, nn = lane & 15;
  for (int r = 0; r < 8; ++r) {
    int m = mt * 16 + r + hi, n = nt * 16 + nn;  // m=k, n=i
    if (m < 24 && n < 24)
      logits[(lb * BB + n) * FK + f * BB + m] = c[r] - cLb[m];
  }
}

// ---------------------------------------------------------------------------
// Kernel 5: softmax over f*k (288) per (h,a,j,i) row -> Lm (f16)
// ---------------------------------------------------------------------------
__global__ __launch_bounds__(FK) void lm_softmax(
    const float* __restrict__ logits, h16* __restrict__ Lm) {
  __shared__ float buf[FK];
  __shared__ float stat[2];
  int row = blockIdx.x, tid = threadIdx.x;
  float v = logits[(size_t)row * FK + tid];
  buf[tid] = v;
  __syncthreads();
  if (tid == 0) {
    float m = -1e30f;
    for (int q = 0; q < FK; ++q) m = fmaxf(m, buf[q]);
    stat[0] = m;
  }
  __syncthreads();
  float e = __expf(v - stat[0]);
  buf[tid] = e;
  __syncthreads();
  if (tid == 0) {
    float s2 = 0.f;
    for (int q = 0; q < FK; ++q) s2 += buf[q];
    stat[1] = 1.f / s2;
  }
  __syncthreads();
  Lm[(size_t)row * FK + tid] = (h16)(e * stat[1]);
}

// ---------------------------------------------------------------------------
// Kernel 6: O[i,d] = Lm(24x288) @ Y(288x128), one block per (h,a,j).
// ---------------------------------------------------------------------------
__global__ __launch_bounds__(128) void attn_combine(
    const h16* __restrict__ Lm, const h16* __restrict__ Ytw, h16* __restrict__ attn) {
  int blk = blockIdx.x;
  int j = blk % BB; int t = blk / BB;
  int a = t % FB; int h = t / FB;
  int tid = threadIdx.x, lane = tid & 31, wave = tid >> 5;
  const h16* A = Lm + ((size_t)((h * FB + a) * BB + j)) * BB * FK;   // [i][fk]
  const h16* B = Ytw + ((size_t)((h * FB + a) * BB + j)) * HD * FK;  // [d][fk]
  int hi = (lane >> 4) << 3, nn = lane & 15;
  for (int q = 0; q < 4; ++q) {
    int ti = wave * 4 + q, mt = ti >> 3, nt = ti & 7;
    v8f c = {};
    for (int kk = 0; kk < 9; ++kk) {  // K = 288 = 9 * 32
      v16h af = frag_a(A + (size_t)(mt * 16) * FK + kk * 32, FK, lane);
      v16h bf = frag_b(B + (size_t)(nt * 16) * FK + kk * 32, FK, lane);
      c = WMMA_F16(af, bf, c);
    }
    for (int r = 0; r < 8; ++r) {
      int m = mt * 16 + r + hi, n = nt * 16 + nn;  // m=i, n=d
      if (m < 24)
        attn[((size_t)((a * BB + m) * BB + j)) * DIM + h * HD + n] = (h16)c[r];
    }
  }
}

// ---------------------------------------------------------------------------
// Host launch
// ---------------------------------------------------------------------------
extern "C" void kernel_launch(void* const* d_in, const int* in_sizes, int n_in,
                              void* d_out, int out_size, void* d_ws, size_t ws_size,
                              hipStream_t stream) {
  (void)in_sizes; (void)n_in; (void)out_size; (void)ws_size;
  const float* x  = (const float*)d_in[0];
  const float* Wq = (const float*)d_in[1];
  const float* bq = (const float*)d_in[2];
  const float* Wk = (const float*)d_in[3];
  const float* bk = (const float*)d_in[4];
  const float* Wv = (const float*)d_in[5];
  const float* bv = (const float*)d_in[6];
  const float* Wo = (const float*)d_in[7];
  const float* bo = (const float*)d_in[8];
  const float* gq = (const float*)d_in[9];
  const float* gk = (const float*)d_in[10];
  const float* fr = (const float*)d_in[11];
  float* out = (float*)d_out;

  char* w = (char*)d_ws;
  size_t off = 0;
  auto alloc = [&](size_t bytes) {
    size_t o = off;
    off += (bytes + 255) & ~(size_t)255;
    return o;
  };
  const size_t SC = (size_t)SEQ * DIM;
  const size_t DD = (size_t)DIM * DIM;
  float* qkvf  = (float*)(w + alloc(3 * SC * 4));
  h16*   xh    = (h16*)(w + alloc(SC * 2));
  h16*   wqh   = (h16*)(w + alloc(DD * 2));
  h16*   wkh   = (h16*)(w + alloc(DD * 2));
  h16*   wvh   = (h16*)(w + alloc(DD * 2));
  h16*   woh   = (h16*)(w + alloc(DD * 2));
  h16*   qhp   = (h16*)(w + alloc(SC * 2));
  h16*   khp   = (h16*)(w + alloc(SC * 2));
  h16*   vhp   = (h16*)(w + alloc(SC * 2));
  h16*   aLw   = (h16*)(w + alloc((size_t)NH * FB * FB * BB * BB * HD * 2));
  float* cLw   = (float*)(w + alloc((size_t)NH * FB * FB * BB * BB * 4));
  h16*   Ytw   = (h16*)(w + alloc((size_t)NH * FB * BB * HD * FK * 2));
  float* logit = (float*)(w + alloc((size_t)NH * FB * BB * BB * FK * 4));
  h16*   Lm    = (h16*)(w + alloc((size_t)NH * FB * BB * BB * FK * 2));
  h16*   attn  = (h16*)(w + alloc(SC * 2));
  (void)alloc(4 << 20);  // slack for padded-row over-reads of the last region

  // One-shot f16 conversions (activations + all weights)
  to_f16<<<4096, 256, 0, stream>>>(x,  xh,  SC);
  to_f16<<<2048, 256, 0, stream>>>(Wq, wqh, DD);
  to_f16<<<2048, 256, 0, stream>>>(Wk, wkh, DD);
  to_f16<<<2048, 256, 0, stream>>>(Wv, wvh, DD);
  to_f16<<<2048, 256, 0, stream>>>(Wo, woh, DD);

  dim3 gg(DIM / 64, SEQ / 128);
  gemm_f16<<<gg, 256, 0, stream>>>(xh, wqh, bq, qkvf,          SEQ, DIM, DIM);
  gemm_f16<<<gg, 256, 0, stream>>>(xh, wkh, bk, qkvf + SC,     SEQ, DIM, DIM);
  gemm_f16<<<gg, 256, 0, stream>>>(xh, wvh, bv, qkvf + 2 * SC, SEQ, DIM, DIM);

  norm_rope<<<SEQ, 256, 0, stream>>>(qkvf,          gq, fr, qhp, 0);
  norm_rope<<<SEQ, 256, 0, stream>>>(qkvf + SC,     gk, fr, khp, 1);
  norm_rope<<<SEQ, 256, 0, stream>>>(qkvf + 2 * SC, gq, fr, vhp, 2);

  right_step<<<NH * FB * FB * BB, 128, 0, stream>>>(qhp, khp, vhp, aLw, cLw, Ytw);
  left_logits<<<NH * FB * FB * BB, 128, 0, stream>>>(aLw, cLw, qhp, logit);
  lm_softmax<<<NH * FB * BB * BB, FK, 0, stream>>>(logit, Lm);
  attn_combine<<<NH * FB * BB, 128, 0, stream>>>(Lm, Ytw, attn);

  gemm_f16<<<gg, 256, 0, stream>>>(attn, woh, bo, out, SEQ, DIM, DIM);
}
